// Block_33964601377062
// MI455X (gfx1250) — matrix operand order repaired and checked by reading, not compiled, not run on hardware
//
#include <hip/hip_runtime.h>
#include <hip/hip_bf16.h>
#include <stdint.h>

#define DIM   768
#define NHEAD 12
#define HD    64
#define SEQ   2048
#define BATCH 2
#define TOK   (BATCH*SEQ)
#define HID   3072
#define QKVN  (3*DIM)

typedef unsigned short u16;
typedef unsigned int   u32;
typedef __attribute__((ext_vector_type(16))) __bf16 bf16x16;
typedef __attribute__((ext_vector_type(8)))  float  f32x8;

struct alignas(16) U4 { u32 x, y, z, w; };
union AF { U4 u[2]; bf16x16 v; };

__device__ __forceinline__ u16 f2bf(float f) {
  u32 u = __float_as_uint(f);
  u32 r = u + 0x7FFFu + ((u >> 16) & 1u);   // round-to-nearest-even
  return (u16)(r >> 16);
}
__device__ __forceinline__ float bf2f(u16 h) {
  return __uint_as_float(((u32)h) << 16);
}

// 16B async copy global -> LDS (ASYNCcnt-tracked, no VGPR data movement).
// VDST = LDS byte offset (low 32 bits of a generic pointer to __shared__),
// VADDR = 64-bit global address, GV mode.
__device__ __forceinline__ void async_ld_b128(void* lds_ptr, const void* gptr) {
  unsigned dst = (unsigned)(unsigned long long)lds_ptr;
  asm volatile("global_load_async_to_lds_b128 %0, %1, off"
               :: "v"(dst), "v"(gptr) : "memory");
}
__device__ __forceinline__ void async_wait0() {
  asm volatile("s_wait_asynccnt 0x0" ::: "memory");
}

// ---------------- fp32 -> bf16 weight conversion (optional per-column scale) --------
__global__ __launch_bounds__(256)
void cvt_bf16_kernel(const float* __restrict__ src, u16* __restrict__ dst,
                     int n, int ncols, int scale_cols, float scale) {
  int i = blockIdx.x * 256 + threadIdx.x;
  if (i >= n) return;
  float v = src[i];
  if ((i % ncols) < scale_cols) v *= scale;
  dst[i] = f2bf(v);
}

// ---------------- RMSNorm: fp32 row -> bf16 row -----------------------------------
__global__ __launch_bounds__(256)
void rmsnorm_bf16_kernel(const float* __restrict__ x, const float* __restrict__ g,
                         u16* __restrict__ out) {
  const int t = blockIdx.x;
  const float* row = x + (size_t)t * DIM;
  const int i0 = threadIdx.x, i1 = threadIdx.x + 256, i2 = threadIdx.x + 512;
  float v0 = row[i0], v1 = row[i1], v2 = row[i2];
  __shared__ float red[256];
  red[threadIdx.x] = v0*v0 + v1*v1 + v2*v2;
  __syncthreads();
  for (int s = 128; s > 0; s >>= 1) {
    if (threadIdx.x < s) red[threadIdx.x] += red[threadIdx.x + s];
    __syncthreads();
  }
  float inv = rsqrtf(red[0] * (1.0f / DIM) + 1e-5f);
  u16* orow = out + (size_t)t * DIM;
  orow[i0] = f2bf(v0 * inv * g[i0]);
  orow[i1] = f2bf(v1 * inv * g[i1]);
  orow[i2] = f2bf(v2 * inv * g[i2]);
}

// ---------------- bf16 WMMA GEMM: C[M,N] = A[M,K] x B[K,N] (+epilogue) --------------
// EPI: 1 = store bf16, 2 = add fp32 residual, store fp32
template<int EPI>
__global__ __launch_bounds__(256)
void gemm_bf16_kernel(const u16* __restrict__ A, const u16* __restrict__ B,
                      const float* __restrict__ resid,
                      float* __restrict__ Cf, u16* __restrict__ Cb,
                      int M, int N, int K) {
  __shared__ u16 ldsA[128 * 32];   // [row][k]
  __shared__ u16 ldsB[128 * 32];   // [n][k]  (transposed at store)
  const int tid  = threadIdx.x;
  const int lane = tid & 31, w = tid >> 5;
  const int lh = lane & 15, hi = lane >> 4;
  const int wr = w >> 2, wc = w & 3;           // wave grid 2x4
  const int m0 = blockIdx.y * 128, n0 = blockIdx.x * 128;

  f32x8 acc[4][2];
  #pragma unroll
  for (int m = 0; m < 4; ++m)
    #pragma unroll
    for (int n = 0; n < 2; ++n)
      #pragma unroll
      for (int v = 0; v < 8; ++v) acc[m][n][v] = 0.0f;

  const int nK = K >> 5;
  for (int kk = 0; kk < nK; ++kk) {
    const int k0 = kk << 5;
    // A tile 128x32: 512 16B chunks, copied global->LDS on the async engine
    #pragma unroll
    for (int i = 0; i < 2; ++i) {
      int c = tid + 256 * i;
      int row = c >> 2, kc = (c & 3) << 3;
      async_ld_b128(&ldsA[row * 32 + kc], &A[(size_t)(m0 + row) * K + k0 + kc]);
    }
    // B tile 32x128, transposed into [n][k] (needs VGPR pack)
    {
      int kp = tid >> 4;              // k pair index 0..15
      int nc = (tid & 15) << 3;       // n chunk base
      U4 r0 = *(const U4*)&B[(size_t)(k0 + kp * 2    ) * N + n0 + nc];
      U4 r1 = *(const U4*)&B[(size_t)(k0 + kp * 2 + 1) * N + n0 + nc];
      u32 a[4] = {r0.x, r0.y, r0.z, r0.w};
      u32 b[4] = {r1.x, r1.y, r1.z, r1.w};
      #pragma unroll
      for (int i = 0; i < 8; ++i) {
        u32 e0 = (a[i >> 1] >> ((i & 1) * 16)) & 0xFFFFu;
        u32 e1 = (b[i >> 1] >> ((i & 1) * 16)) & 0xFFFFu;
        *(u32*)&ldsB[(nc + i) * 32 + kp * 2] = e0 | (e1 << 16);
      }
    }
    if (kk + 1 < nK) {  // warm next tiles -> global_prefetch_b8
      __builtin_prefetch(&A[(size_t)(m0 + (tid >> 1)) * K + k0 + 32], 0, 1);
      __builtin_prefetch(&B[(size_t)(k0 + 32 + (tid >> 4)) * N + n0 + (tid & 15) * 8], 0, 1);
    }
    async_wait0();
    __syncthreads();

    AF aF[4], bF[2];
    #pragma unroll
    for (int m = 0; m < 4; ++m) {
      int row = wr * 64 + m * 16 + lh;
      aF[m].u[0] = *(const U4*)&ldsA[row * 32 + hi * 8];        // K = hi*8 .. +7
      aF[m].u[1] = *(const U4*)&ldsA[row * 32 + 16 + hi * 8];   // K = 16+hi*8 ..
    }
    #pragma unroll
    for (int n = 0; n < 2; ++n) {
      int col = wc * 32 + n * 16 + lh;
      bF[n].u[0] = *(const U4*)&ldsB[col * 32 + hi * 8];        // K = hi*8 .. +7
      bF[n].u[1] = *(const U4*)&ldsB[col * 32 + 16 + hi * 8];   // K = 16+hi*8 ..
    }
    #pragma unroll
    for (int m = 0; m < 4; ++m)
      #pragma unroll
      for (int n = 0; n < 2; ++n)
        acc[m][n] = __builtin_amdgcn_wmma_f32_16x16x32_bf16(
            false, aF[m].v, false, bF[n].v, (short)0, acc[m][n], false, false);
    __syncthreads();
  }

  // Epilogue: D layout => M = v + 8*hi, N = lh within each 16x16 fragment
  #pragma unroll
  for (int m = 0; m < 4; ++m)
    #pragma unroll
    for (int n = 0; n < 2; ++n)
      #pragma unroll
      for (int v = 0; v < 8; ++v) {
        int row = m0 + wr * 64 + m * 16 + hi * 8 + v;
        int col = n0 + wc * 32 + n * 16 + lh;
        size_t idx = (size_t)row * N + col;
        float val = acc[m][n][v];
        if (EPI == 1)      Cb[idx] = f2bf(val);
        else               Cf[idx] = val + resid[idx];
      }
}

// ---------------- flash attention, one (batch, head, 64-q-row) block ----------------
__global__ __launch_bounds__(128)
void attn_kernel(const u16* __restrict__ qkv, u16* __restrict__ out) {
  __shared__ u16 ldsK[64 * 64];      // [key][hd]
  __shared__ u16 ldsV[64 * 64];      // [hd][key]  (transposed)
  __shared__ u16 ldsP[4][16 * 32];   // per-wave probs
  const int tid  = threadIdx.x;
  const int lane = tid & 31, w = tid >> 5;     // 4 waves, 16 q rows each
  const int lh = lane & 15, hi = lane >> 4;
  const int qb = blockIdx.x, h = blockIdx.y, b = blockIdx.z;
  const size_t base = (size_t)b * SEQ;
  const int qcol = h * HD, kcol = DIM + h * HD, vcol = 2 * DIM + h * HD;

  // Q fragments live in registers for the whole kernel (hd=64 => 2 A-frags)
  AF qF[2];
  {
    int t = qb * 64 + w * 16 + lh;
    const u16* qp = qkv + (base + t) * QKVN + qcol;
    qF[0].u[0] = *(const U4*)&qp[hi * 8];
    qF[0].u[1] = *(const U4*)&qp[16 + hi * 8];
    qF[1].u[0] = *(const U4*)&qp[32 + hi * 8];
    qF[1].u[1] = *(const U4*)&qp[48 + hi * 8];
  }

  float mrow[8], lrow[8];
  f32x8 o[4];
  #pragma unroll
  for (int v = 0; v < 8; ++v) { mrow[v] = -1e30f; lrow[v] = 0.0f; }
  #pragma unroll
  for (int n = 0; n < 4; ++n)
    #pragma unroll
    for (int v = 0; v < 8; ++v) o[n][v] = 0.0f;

  for (int kc0 = 0; kc0 < SEQ; kc0 += 64) {
    // K tile: straight row-major copy -> async engine; V tile: transpose via VGPRs
    #pragma unroll
    for (int i = 0; i < 4; ++i) {
      int c = tid + 128 * i;               // 0..511
      int row = c >> 3, hc = (c & 7) << 3;
      const u16* kp = qkv + (base + kc0 + row) * QKVN;
      async_ld_b128(&ldsK[row * 64 + hc], &kp[kcol + hc]);
      U4 vv = *(const U4*)&kp[vcol + hc];
      u32 e[4] = {vv.x, vv.y, vv.z, vv.w};
      #pragma unroll
      for (int j = 0; j < 8; ++j)
        ldsV[(hc + j) * 64 + row] = (u16)((e[j >> 1] >> ((j & 1) * 16)) & 0xFFFFu);
    }
    async_wait0();
    __syncthreads();

    #pragma unroll
    for (int kg = 0; kg < 2; ++kg) {       // two 32-key groups per chunk
      const int kb = kg * 32;
      f32x8 s0, s1;
      #pragma unroll
      for (int v = 0; v < 8; ++v) { s0[v] = 0.0f; s1[v] = 0.0f; }
      {
        AF bf;
        const int kr0 = (kb + lh) * 64;          // keys kb..kb+15
        bf.u[0] = *(const U4*)&ldsK[kr0 + hi * 8];
        bf.u[1] = *(const U4*)&ldsK[kr0 + 16 + hi * 8];
        s0 = __builtin_amdgcn_wmma_f32_16x16x32_bf16(false, qF[0].v, false, bf.v, (short)0, s0, false, false);
        bf.u[0] = *(const U4*)&ldsK[kr0 + 32 + hi * 8];
        bf.u[1] = *(const U4*)&ldsK[kr0 + 48 + hi * 8];
        s0 = __builtin_amdgcn_wmma_f32_16x16x32_bf16(false, qF[1].v, false, bf.v, (short)0, s0, false, false);
        const int kr1 = (kb + 16 + lh) * 64;     // keys kb+16..kb+31
        bf.u[0] = *(const U4*)&ldsK[kr1 + hi * 8];
        bf.u[1] = *(const U4*)&ldsK[kr1 + 16 + hi * 8];
        s1 = __builtin_amdgcn_wmma_f32_16x16x32_bf16(false, qF[0].v, false, bf.v, (short)0, s1, false, false);
        bf.u[0] = *(const U4*)&ldsK[kr1 + 32 + hi * 8];
        bf.u[1] = *(const U4*)&ldsK[kr1 + 48 + hi * 8];
        s1 = __builtin_amdgcn_wmma_f32_16x16x32_bf16(false, qF[1].v, false, bf.v, (short)0, s1, false, false);
      }
      // online softmax: each lane owns rows v+8*hi; 16-lane shfl_xor reductions
      #pragma unroll
      for (int v = 0; v < 8; ++v) {
        float a = s0[v], c = s1[v];
        float mx = fmaxf(a, c);
        #pragma unroll
        for (int off = 1; off < 16; off <<= 1) mx = fmaxf(mx, __shfl_xor(mx, off, 32));
        float mnew = fmaxf(mrow[v], mx);
        float corr = __expf(mrow[v] - mnew);
        float p0 = __expf(a - mnew);
        float p1 = __expf(c - mnew);
        float rs = p0 + p1;
        #pragma unroll
        for (int off = 1; off < 16; off <<= 1) rs += __shfl_xor(rs, off, 32);
        lrow[v] = lrow[v] * corr + rs;
        mrow[v] = mnew;
        #pragma unroll
        for (int n = 0; n < 4; ++n) o[n][v] *= corr;
        ldsP[w][(hi * 8 + v) * 32 + lh]      = f2bf(p0);
        ldsP[w][(hi * 8 + v) * 32 + 16 + lh] = f2bf(p1);
      }
      // P (16x32) x V (32x64): read P back in A-fragment layout
      AF pF;
      pF.u[0] = *(const U4*)&ldsP[w][lh * 32 + hi * 8];
      pF.u[1] = *(const U4*)&ldsP[w][lh * 32 + 16 + hi * 8];
      #pragma unroll
      for (int n = 0; n < 4; ++n) {
        AF vF;
        vF.u[0] = *(const U4*)&ldsV[(n * 16 + lh) * 64 + kb + hi * 8];
        vF.u[1] = *(const U4*)&ldsV[(n * 16 + lh) * 64 + kb + 16 + hi * 8];
        o[n] = __builtin_amdgcn_wmma_f32_16x16x32_bf16(false, pF.v, false, vF.v, (short)0, o[n], false, false);
      }
    }
    __syncthreads();
  }
  // normalize and write bf16 attention output
  #pragma unroll
  for (int n = 0; n < 4; ++n)
    #pragma unroll
    for (int v = 0; v < 8; ++v) {
      int t = qb * 64 + w * 16 + hi * 8 + v;
      out[(base + t) * DIM + qcol + n * 16 + lh] = f2bf(o[n][v] / lrow[v]);
    }
}

// ---------------- GeGLU-mish: a = mish(h1) * gate ----------------------------------
__global__ __launch_bounds__(256)
void geglu_mish_kernel(const u16* __restrict__ hg, u16* __restrict__ outa) {
  size_t i = (size_t)blockIdx.x * 256 + threadIdx.x;
  if (i >= (size_t)TOK * HID) return;
  size_t row = i / HID, c = i % HID;
  const u16* p = hg + row * (size_t)(2 * HID);
  float x = bf2f(p[c]);
  float g = bf2f(p[HID + c]);
  float sp = (x > 15.0f) ? x : log1pf(__expf(x));
  outa[i] = f2bf(x * tanhf(sp) * g);
}

extern "C" void kernel_launch(void* const* d_in, const int* in_sizes, int n_in,
                              void* d_out, int out_size, void* d_ws, size_t ws_size,
                              hipStream_t stream) {
  (void)in_sizes; (void)n_in; (void)out_size; (void)ws_size;
  const float* x      = (const float*)d_in[0];
  const float* w_qkv  = (const float*)d_in[1];
  const float* w_out  = (const float*)d_in[2];
  const float* w1     = (const float*)d_in[3];
  const float* w2     = (const float*)d_in[4];
  const float* g_attn = (const float*)d_in[5];
  const float* g_ff   = (const float*)d_in[6];

  char* ws = (char*)d_ws;
  size_t off = 0;
  auto alloc = [&](size_t bytes) -> void* {
    void* p = ws + off;
    off = (off + bytes + 255) & ~(size_t)255;
    return p;
  };
  u16*   wqkvb = (u16*)alloc((size_t)DIM * QKVN * 2);
  u16*   woutb = (u16*)alloc((size_t)DIM * DIM * 2);
  u16*   w1b   = (u16*)alloc((size_t)DIM * 2 * HID * 2);
  u16*   w2b   = (u16*)alloc((size_t)HID * DIM * 2);
  u16*   xb    = (u16*)alloc((size_t)TOK * DIM * 2);
  u16*   qkvb  = (u16*)alloc((size_t)TOK * QKVN * 2);
  u16*   attnb = (u16*)alloc((size_t)TOK * DIM * 2);
  float* x1    = (float*)alloc((size_t)TOK * DIM * 4);
  u16*   hffb  = (u16*)alloc((size_t)TOK * DIM * 2);
  u16*   hgb   = (u16*)alloc((size_t)TOK * 2 * HID * 2);
  u16*   affb  = (u16*)alloc((size_t)TOK * HID * 2);

  // weights -> bf16; fold 1/sqrt(hd)=0.125 into q AND k columns (per reference)
  cvt_bf16_kernel<<<(DIM*QKVN + 255)/256, 256, 0, stream>>>(w_qkv, wqkvb, DIM*QKVN, QKVN, 2*DIM, 0.125f);
  cvt_bf16_kernel<<<(DIM*DIM + 255)/256, 256, 0, stream>>>(w_out, woutb, DIM*DIM, DIM, 0, 1.0f);
  cvt_bf16_kernel<<<(DIM*2*HID + 255)/256, 256, 0, stream>>>(w1, w1b, DIM*2*HID, 2*HID, 0, 1.0f);
  cvt_bf16_kernel<<<(HID*DIM + 255)/256, 256, 0, stream>>>(w2, w2b, HID*DIM, DIM, 0, 1.0f);

  // attention sub-block
  rmsnorm_bf16_kernel<<<TOK, 256, 0, stream>>>(x, g_attn, xb);
  gemm_bf16_kernel<1><<<dim3(QKVN/128, TOK/128), 256, 0, stream>>>(xb, wqkvb, nullptr, nullptr, qkvb, TOK, QKVN, DIM);
  attn_kernel<<<dim3(SEQ/64, NHEAD, BATCH), 128, 0, stream>>>(qkvb, attnb);
  gemm_bf16_kernel<2><<<dim3(DIM/128, TOK/128), 256, 0, stream>>>(attnb, woutb, x, x1, nullptr, TOK, DIM, DIM);

  // FFN sub-block
  rmsnorm_bf16_kernel<<<TOK, 256, 0, stream>>>(x1, g_ff, hffb);
  gemm_bf16_kernel<1><<<dim3(2*HID/128, TOK/128), 256, 0, stream>>>(hffb, w1b, nullptr, nullptr, hgb, TOK, 2*HID, DIM);
  geglu_mish_kernel<<<(unsigned)(((size_t)TOK*HID + 255)/256), 256, 0, stream>>>(hgb, affb);
  gemm_bf16_kernel<2><<<dim3(DIM/128, TOK/128), 256, 0, stream>>>(affb, w2b, x1, (float*)d_out, nullptr, TOK, DIM, HID);
}
